// GNN_LSTM_Model_23622320128293
// MI455X (gfx1250) — matrix operand order repaired and checked by reading, not compiled, not run on hardware
//
#include <hip/hip_runtime.h>

#define DFEAT 256
#define HID   512
#define GATE  2048
#define NWG   8          // workgroups cooperating on the LSTM recurrence

typedef __bf16 v16bf __attribute__((ext_vector_type(16)));
typedef float  v8f   __attribute__((ext_vector_type(8)));

union BFrag { v16bf v; uint4 u[2]; };
union FAcc  { v8f v; float f[8]; };

__device__ __forceinline__ unsigned short f2bf(float f) {
    unsigned int u = __float_as_uint(f);
    u += 0x7FFFu + ((u >> 16) & 1u);            // round-to-nearest-even
    return (unsigned short)(u >> 16);
}
__device__ __forceinline__ float bf2f(unsigned short s) {
    return __uint_as_float(((unsigned int)s) << 16);
}

// ---------------- scatter: hsum (pre-initialized to x) += x[src] at row dst ----
__global__ void k_scatter(const float* __restrict__ x, const int* __restrict__ ei,
                          float* __restrict__ hsum, int E)
{
    long idx = (long)blockIdx.x * blockDim.x + threadIdx.x;
    if (idx >= (long)E * (DFEAT / 4)) return;
    int e = (int)(idx / (DFEAT / 4));
    int f = (int)(idx % (DFEAT / 4)) << 2;
    int s = ei[e], d = ei[E + e];
    const float4 val = *(const float4*)(x + (long)s * DFEAT + f);
    float* dp = hsum + (long)d * DFEAT + f;
    atomicAdd(dp + 0, val.x); atomicAdd(dp + 1, val.y);
    atomicAdd(dp + 2, val.z); atomicAdd(dp + 3, val.w);
}

// ---------------- f32 -> bf16 bulk convert ------------------------------------
__global__ void k_cvt_bf16(const float* __restrict__ s, unsigned short* __restrict__ d, long n)
{
    long i = (long)blockIdx.x * blockDim.x + threadIdx.x;
    if (i < n) d[i] = f2bf(s[i]);
}

// ------- pack recurrent weights: [4H][H] f32 -> [H/8][4H] uint4 (8 x bf16) -----
__global__ void k_pack_whh(const float* __restrict__ w, uint4* __restrict__ wq)
{
    int i = blockIdx.x * blockDim.x + threadIdx.x;   // (HID/8) * GATE
    if (i >= (HID / 8) * GATE) return;
    int kb = i / GATE, r = i % GATE;
    const float* base = w + (long)r * HID + kb * 8;
    uint4 q;
    q.x = (unsigned)f2bf(base[0]) | ((unsigned)f2bf(base[1]) << 16);
    q.y = (unsigned)f2bf(base[2]) | ((unsigned)f2bf(base[3]) << 16);
    q.z = (unsigned)f2bf(base[4]) | ((unsigned)f2bf(base[5]) << 16);
    q.w = (unsigned)f2bf(base[6]) | ((unsigned)f2bf(base[7]) << 16);
    wq[(long)kb * GATE + r] = q;
}

__global__ void k_init_cnt(unsigned int* c) { if (threadIdx.x < 2) c[threadIdx.x] = 0u; }

// ---------------- WMMA GEMM: C[M,Nc] = A[M,K](bf16) * B[Nc,K](bf16)^T ---------
__global__ void __launch_bounds__(256)
k_gemm_wmma(const unsigned short* __restrict__ A, const unsigned short* __restrict__ B,
            float* __restrict__ Cf, unsigned short* __restrict__ Cb,
            const float* __restrict__ bias1, const float* __restrict__ bias2,
            int M, int Nc, int K, int relu)
{
    const int lane = threadIdx.x & 31;
    const int wave = threadIdx.x >> 5;
    const int tnc  = Nc >> 4;
    const long tile = (long)blockIdx.x * 8 + wave;
    if (tile >= (long)(M >> 4) * tnc) return;       // wave-uniform: EXEC stays full
    const int tm = (int)(tile / tnc);
    const int tn = (int)(tile % tnc);
    const int l15  = lane & 15;
    const int ksel = lane >> 4;

    const unsigned short* ap = A + (long)((tm << 4) + l15) * K + (ksel << 3);
    const unsigned short* bp = B + (long)((tn << 4) + l15) * K + (ksel << 3);

    FAcc acc;
#pragma unroll
    for (int i = 0; i < 8; ++i) acc.f[i] = 0.f;

    for (int kk = 0; kk < K; kk += 32) {
        BFrag a, b;
        a.u[0] = *(const uint4*)(ap);
        a.u[1] = *(const uint4*)(ap + 16);
        b.u[0] = *(const uint4*)(bp);
        b.u[1] = *(const uint4*)(bp + 16);
        ap += 32; bp += 32;
        __builtin_prefetch(ap, 0, 1);               // global_prefetch_b8
        __builtin_prefetch(bp, 0, 1);
        acc.v = __builtin_amdgcn_wmma_f32_16x16x32_bf16(
            false, a.v, false, b.v, (short)0, acc.v, false, false);
    }

    const int col = (tn << 4) + l15;
    float bv = 0.f;
    if (bias1) bv += bias1[col];
    if (bias2) bv += bias2[col];
#pragma unroll
    for (int v = 0; v < 8; ++v) {
        int rowo = (tm << 4) + (ksel << 3) + v;
        float val = acc.f[v] + bv;
        if (relu) val = fmaxf(val, 0.f);
        long idx = (long)rowo * Nc + col;
        if (Cf) Cf[idx] = val;
        if (Cb) Cb[idx] = f2bf(val);
    }
}

// ---------------- BatchNorm stats / apply -------------------------------------
__global__ void k_bn_stats(const float* __restrict__ h, float* __restrict__ mean,
                           float* __restrict__ rstd, int M)
{
    __shared__ float s1[256], s2[256];
    int c = blockIdx.x;
    float a = 0.f, b = 0.f;
    for (int r = threadIdx.x; r < M; r += 256) {
        float v = h[(long)r * HID + c];
        a += v; b += v * v;
    }
    s1[threadIdx.x] = a; s2[threadIdx.x] = b;
    __syncthreads();
    for (int s = 128; s > 0; s >>= 1) {
        if (threadIdx.x < s) { s1[threadIdx.x] += s1[threadIdx.x + s];
                               s2[threadIdx.x] += s2[threadIdx.x + s]; }
        __syncthreads();
    }
    if (threadIdx.x == 0) {
        float m = s1[0] / (float)M;
        float v = s2[0] / (float)M - m * m;
        mean[c] = m;
        rstd[c] = rsqrtf(v + 1e-5f);
    }
}

__global__ void k_bn_apply(const float* __restrict__ h, const float* __restrict__ mean,
                           const float* __restrict__ rstd, const float* __restrict__ gamma,
                           const float* __restrict__ beta, unsigned short* __restrict__ hb, long n)
{
    long i = (long)blockIdx.x * blockDim.x + threadIdx.x;
    if (i >= n) return;
    int c = (int)(i & (HID - 1));
    float v = (h[i] - mean[c]) * rstd[c] * gamma[c] + beta[c];
    hb[i] = f2bf(fmaxf(v, 0.f));
}

// ------------- LSTM recurrence split across NWG workgroups --------------------
// Each WG owns 64 h-lanes (256 gate rows). Per step: gate GEMV slice (4 threads
// per row over k-quarters), LDS reduce, c/h update, publish h slice to a
// double-buffered global array, then agent-scope atomic split-barrier.
// s_cluster_barrier is emitted on the arrival path (architectural NOP when the
// grid is not dispatched as a cluster); the atomic protocol is the real sync.
__global__ void __launch_bounds__(1024)
k_lstm_par(const float* __restrict__ pre, const uint4* __restrict__ wq,
           unsigned short* __restrict__ hsb, float* __restrict__ hsf,
           float* __restrict__ hglob,          // [2][HID] exchange buffer
           unsigned int* __restrict__ cnt,     // arrival counter (pre-zeroed)
           float* __restrict__ hn, float* __restrict__ cn, int T)
{
    __shared__ float hs[HID];
    __shared__ float partial[1024];
    __shared__ float gs[256];
    const int tid  = threadIdx.x;
    const int wg   = blockIdx.x;            // 0..NWG-1
    const int sub  = tid >> 8;              // k-quarter 0..3
    const int rl   = tid & 255;
    const int gate = rl >> 6;               // i,f,g,o group
    const int jo   = rl & 63;
    const int row  = gate * HID + wg * 64 + jo;   // gate row in [0,4H)
    float c = 0.f;
    if (tid < HID) hs[tid] = 0.f;
    __syncthreads();
    const uint4* wbase = wq + row;
    const float4* h4 = (const float4*)hs;

    for (int t = 0; t < T; ++t) {
        float pv = 0.f;
        if (tid < 256) pv = pre[(long)t * GATE + row];   // hoisted early load
        float a = 0.f;
#pragma unroll
        for (int kb = sub * 16; kb < sub * 16 + 16; ++kb) {
            uint4 q = wbase[(long)kb * GATE];            // 8 bf16 weights
            float4 ha = h4[2 * kb];
            float4 hb = h4[2 * kb + 1];
            a = fmaf(ha.x, bf2f((unsigned short)q.x), a);
            a = fmaf(ha.y, bf2f((unsigned short)(q.x >> 16)), a);
            a = fmaf(ha.z, bf2f((unsigned short)q.y), a);
            a = fmaf(ha.w, bf2f((unsigned short)(q.y >> 16)), a);
            a = fmaf(hb.x, bf2f((unsigned short)q.z), a);
            a = fmaf(hb.y, bf2f((unsigned short)(q.z >> 16)), a);
            a = fmaf(hb.z, bf2f((unsigned short)q.w), a);
            a = fmaf(hb.w, bf2f((unsigned short)(q.w >> 16)), a);
        }
        partial[tid] = a;
        __syncthreads();
        if (tid < 256)
            gs[tid] = pv + partial[tid] + partial[256 + tid]
                         + partial[512 + tid] + partial[768 + tid];
        __syncthreads();
        if (tid < 64) {                                 // torch gate order i,f,g,o
            float ig = 1.f / (1.f + expf(-gs[tid]));
            float fg = 1.f / (1.f + expf(-gs[64 + tid]));
            float gg = tanhf(gs[128 + tid]);
            float og = 1.f / (1.f + expf(-gs[192 + tid]));
            c = fmaf(fg, c, ig * gg);
            float h = og * tanhf(c);
            int gi = wg * 64 + tid;
            hglob[(t & 1) * HID + gi] = h;
            if (hsb) hsb[(long)t * HID + gi] = f2bf(h);
            if (hsf) hsf[(long)t * HID + gi] = h;
        }
        __threadfence();                                // release h slice (agent)
        __syncthreads();
        if (tid == 0) {
            __builtin_amdgcn_s_cluster_barrier();       // NOP unless clustered
            __hip_atomic_fetch_add(cnt, 1u, __ATOMIC_RELEASE, __HIP_MEMORY_SCOPE_AGENT);
            unsigned int target = (unsigned)NWG * (unsigned)(t + 1);
            while (__hip_atomic_load(cnt, __ATOMIC_ACQUIRE, __HIP_MEMORY_SCOPE_AGENT) < target)
                __builtin_amdgcn_s_sleep(4);
        }
        __syncthreads();
        __threadfence();                                // acquire before reading peers
        if (tid < HID) hs[tid] = hglob[(t & 1) * HID + tid];
        __syncthreads();
    }
    if (tid < 64) {
        hn[wg * 64 + tid] = hs[wg * 64 + tid];          // hs holds final h
        cn[wg * 64 + tid] = c;
    }
}

// ---------------- FC head: one wave per row, shuffle reduction ----------------
__global__ void __launch_bounds__(256)
k_fc(const float* __restrict__ hs1, const float* __restrict__ fcw,
     const float* __restrict__ fcb, float* __restrict__ out, int T)
{
    int row  = blockIdx.x * 8 + (threadIdx.x >> 5);
    int lane = threadIdx.x & 31;
    if (row >= T) return;
    float a = 0.f;
    for (int k = lane; k < HID; k += 32)
        a = fmaf(hs1[(long)row * HID + k], fcw[k], a);
    for (int off = 16; off > 0; off >>= 1) a += __shfl_down(a, off, 32);
    if (lane == 0) out[row] = a + fcb[0];
}

extern "C" void kernel_launch(void* const* d_in, const int* in_sizes, int n_in,
                              void* d_out, int out_size, void* d_ws, size_t ws_size,
                              hipStream_t stream) {
    const float* x     = (const float*)d_in[0];
    const int*   ei    = (const int*)d_in[1];
    const float* w1    = (const float*)d_in[2];
    const float* b1    = (const float*)d_in[3];
    const float* w2    = (const float*)d_in[4];
    const float* b2    = (const float*)d_in[5];
    const float* gamma = (const float*)d_in[6];
    const float* beta  = (const float*)d_in[7];
    const float* w_ih0 = (const float*)d_in[8];
    const float* w_hh0 = (const float*)d_in[9];
    const float* b_ih0 = (const float*)d_in[10];
    const float* b_hh0 = (const float*)d_in[11];
    const float* w_ih1 = (const float*)d_in[12];
    const float* w_hh1 = (const float*)d_in[13];
    const float* b_ih1 = (const float*)d_in[14];
    const float* b_hh1 = (const float*)d_in[15];
    const float* fc_w  = (const float*)d_in[16];
    const float* fc_b  = (const float*)d_in[17];
    float* out = (float*)d_out;

    const int N = in_sizes[0] / DFEAT;   // 10000 (multiple of 16)
    const int E = in_sizes[1] / 2;       // 640000

    char* ws = (char*)d_ws;
    size_t off = 0;
    auto take = [&](size_t bytes) -> char* {
        char* p = ws + off;
        off = (off + bytes + 255) & ~(size_t)255;
        return p;
    };
    float*          hsum  = (float*)take((size_t)N * DFEAT * 4);
    unsigned short* xb    = (unsigned short*)take((size_t)N * DFEAT * 2);
    unsigned short* w1b   = (unsigned short*)take((size_t)HID * DFEAT * 2);
    unsigned short* w2b   = (unsigned short*)take((size_t)HID * HID * 2);
    unsigned short* wih0b = (unsigned short*)take((size_t)GATE * HID * 2);
    unsigned short* wih1b = (unsigned short*)take((size_t)GATE * HID * 2);
    uint4*          whh0q = (uint4*)take((size_t)(HID / 8) * GATE * 16);
    uint4*          whh1q = (uint4*)take((size_t)(HID / 8) * GATE * 16);
    unsigned short* mid   = (unsigned short*)take((size_t)N * HID * 2);
    float*          hfull = (float*)take((size_t)N * HID * 4);
    float*          mean  = (float*)take((size_t)HID * 4);
    float*          rstd  = (float*)take((size_t)HID * 4);
    unsigned short* hb    = (unsigned short*)take((size_t)N * HID * 2);
    float*          pre   = (float*)take((size_t)N * GATE * 4);   // reused layer0/1
    unsigned short* hs0b  = (unsigned short*)take((size_t)N * HID * 2);
    float*          hs1   = (float*)take((size_t)N * HID * 4);
    float*          hglob = (float*)take((size_t)2 * HID * 4);
    unsigned int*   cnt   = (unsigned int*)take(256);

    // 0) zero the split-barrier counters (ws is NOT re-initialized between calls)
    k_init_cnt<<<1, 32, 0, stream>>>(cnt);

    // 1) GIN aggregation: hsum = x + segment_sum(x[src] -> dst)
    hipMemcpyAsync(hsum, x, (size_t)N * DFEAT * 4, hipMemcpyDeviceToDevice, stream);
    {
        long work = (long)E * (DFEAT / 4);
        k_scatter<<<(int)((work + 255) / 256), 256, 0, stream>>>(x, ei, hsum, E);
    }

    // 2) bf16 staging of activations and weights
    auto cvt = [&](const float* s, unsigned short* d, long n) {
        k_cvt_bf16<<<(int)((n + 255) / 256), 256, 0, stream>>>(s, d, n);
    };
    cvt(hsum,  xb,    (long)N * DFEAT);
    cvt(w1,    w1b,   (long)HID * DFEAT);
    cvt(w2,    w2b,   (long)HID * HID);
    cvt(w_ih0, wih0b, (long)GATE * HID);
    cvt(w_ih1, wih1b, (long)GATE * HID);
    {
        int n = (HID / 8) * GATE;
        k_pack_whh<<<(n + 255) / 256, 256, 0, stream>>>(w_hh0, whh0q);
        k_pack_whh<<<(n + 255) / 256, 256, 0, stream>>>(w_hh1, whh1q);
    }

    // 3) GIN MLP: relu(hsum @ w1^T + b1) @ w2^T + b2
    long tilesH = (long)(N / 16) * (HID / 16);
    k_gemm_wmma<<<(int)((tilesH + 7) / 8), 256, 0, stream>>>(
        xb, w1b, nullptr, mid, b1, nullptr, N, HID, DFEAT, 1);
    k_gemm_wmma<<<(int)((tilesH + 7) / 8), 256, 0, stream>>>(
        mid, w2b, hfull, nullptr, b2, nullptr, N, HID, HID, 0);

    // 4) BatchNorm (training stats) + ReLU -> bf16
    k_bn_stats<<<HID, 256, 0, stream>>>(hfull, mean, rstd, N);
    k_bn_apply<<<(int)(((long)N * HID + 255) / 256), 256, 0, stream>>>(
        hfull, mean, rstd, gamma, beta, hb, (long)N * HID);

    // 5) LSTM layer 0
    long tilesG = (long)(N / 16) * (GATE / 16);
    k_gemm_wmma<<<(int)((tilesG + 7) / 8), 256, 0, stream>>>(
        hb, wih0b, pre, nullptr, b_ih0, b_hh0, N, GATE, HID, 0);
    k_lstm_par<<<NWG, 1024, 0, stream>>>(pre, whh0q, hs0b, nullptr, hglob, cnt,
                                         out + N, out + N + 2 * HID, N);

    // 6) LSTM layer 1 (reuse `pre`)
    k_gemm_wmma<<<(int)((tilesG + 7) / 8), 256, 0, stream>>>(
        hs0b, wih1b, pre, nullptr, b_ih1, b_hh1, N, GATE, HID, 0);
    k_lstm_par<<<NWG, 1024, 0, stream>>>(pre, whh1q, nullptr, hs1, hglob, cnt + 1,
                                         out + N + HID, out + N + 3 * HID, N);

    // 7) FC head -> out[0..N)
    k_fc<<<(N + 7) / 8, 256, 0, stream>>>(hs1, fc_w, fc_b, out, N);
}
